// SioConvPSBlock_84808424226875
// MI455X (gfx1250) — matrix operand order repaired and checked by reading, not compiled, not run on hardware
//
#include <hip/hip_runtime.h>
#include <stdint.h>

// ---------------- problem constants ----------------
constexpr int Bb   = 4;
constexpr int Ll   = 4096;
constexpr int Dd   = 1024;
constexpr int FFf  = 4096;
constexpr int Mtot = Bb * Ll;          // 16384 rows
constexpr int CHUNKS = 32;             // scan chunks per sequence
constexpr int LC     = Ll / CHUNKS;    // 128
constexpr int CHM    = 4096;           // FFN row-chunk

#define DEV __device__ __forceinline__
#define NEG_INF (-__builtin_inff())

typedef __attribute__((ext_vector_type(16))) __bf16 v16bf;
typedef __attribute__((ext_vector_type(8)))  float  v8f;

// ---------------- helpers ----------------
DEV unsigned short f2bf(float f) {                // RNE fp32 -> bf16
  uint32_t u = __builtin_bit_cast(uint32_t, f);
  u += 0x7FFFu + ((u >> 16) & 1u);
  return (unsigned short)(u >> 16);
}
DEV float bf2f(unsigned short h) {
  uint32_t u = ((uint32_t)h) << 16;
  return __builtin_bit_cast(float, u);
}
DEV float softplusf(float t) {                    // stable softplus
  return fmaxf(t, 0.f) + log1pf(__expf(-fabsf(t)));
}
DEV float lae(float a, float b) {                 // logaddexp, -inf safe
  float m = fmaxf(a, b);
  if (m == NEG_INF) return NEG_INF;
  return m + log1pf(__expf(fminf(a, b) - m));
}
DEV float siluf(float t) { return t / (1.f + __expf(-t)); }

// ---------------- RMSNorm -> bf16 ----------------
__global__ __launch_bounds__(256)
void rmsnorm_bf16_kernel(const float* __restrict__ x, const float* __restrict__ g,
                         unsigned short* __restrict__ out) {
  const int row = blockIdx.x;
  const int t   = threadIdx.x;
  const float* xr = x + (size_t)row * Dd;
  float v[4]; float s = 0.f;
#pragma unroll
  for (int i = 0; i < 4; ++i) { v[i] = xr[t + 256 * i]; s += v[i] * v[i]; }
  __shared__ float red[256];
  red[t] = s; __syncthreads();
  for (int off = 128; off > 0; off >>= 1) {
    if (t < off) red[t] += red[t + off];
    __syncthreads();
  }
  const float rms = rsqrtf(red[0] * (1.f / Dd) + 1e-6f);
#pragma unroll
  for (int i = 0; i < 4; ++i)
    out[(size_t)row * Dd + t + 256 * i] = f2bf(v[i] * rms * g[t + 256 * i]);
}

// ---------------- chunked scan (cumsum + cumlogsumexp) ----------------
// thread layout: tid = ((b*CHUNKS + c) << 10) + d
__global__ __launch_bounds__(256)
void scan_p1(const float* __restrict__ zpre, const float* __restrict__ dtb,
             float* __restrict__ csum, float* __restrict__ clse) {
  const int tid = blockIdx.x * 256 + threadIdx.x;
  const int d = tid & (Dd - 1);
  const int bc = tid >> 10;
  const int c = bc & (CHUNKS - 1);
  const int b = bc >> 5;
  size_t idx = ((size_t)(b * Ll + c * LC)) * Dd + d;
  float cum = 0.f, lse = NEG_INF;
  for (int i = 0; i < LC; ++i) {
    float zp = zpre[idx], dv = dtb[idx];
    float ln_z  = -softplusf(-zp);
    float ln_da = -softplusf(-dv);
    float ln_o  = -softplusf(dv);
    cum += ln_o;
    lse = lae(lse, ln_z + ln_da - cum);
    idx += Dd;
  }
  csum[tid] = cum; clse[tid] = lse;
}

__global__ __launch_bounds__(256)
void scan_p2(const float* __restrict__ csum, const float* __restrict__ clse,
             float* __restrict__ Pexc, float* __restrict__ Rexc) {
  const int tid = blockIdx.x * 256 + threadIdx.x;   // 0..B*D-1
  const int d = tid & (Dd - 1);
  const int b = tid >> 10;
  float P = 0.f, R = NEG_INF;
  for (int c = 0; c < CHUNKS; ++c) {
    const int j = ((b * CHUNKS + c) << 10) + d;
    float s = csum[j], l = clse[j];
    Pexc[j] = P; Rexc[j] = R;
    R = lae(R, l - P);
    P += s;
  }
}

__global__ __launch_bounds__(256)
void scan_p3(const float* __restrict__ zpre, const float* __restrict__ dtb,
             const float* __restrict__ Pexc, const float* __restrict__ Rexc,
             const float* __restrict__ hidden,
             float* __restrict__ outH, unsigned short* __restrict__ h_bf) {
  const int tid = blockIdx.x * 256 + threadIdx.x;
  const int d = tid & (Dd - 1);
  const int bc = tid >> 10;
  const int c = bc & (CHUNKS - 1);
  const int b = bc >> 5;
  const float P = Pexc[tid], R = Rexc[tid];
  const float hid = hidden[b * Dd + d];
  size_t idx = ((size_t)(b * Ll + c * LC)) * Dd + d;
  float cum = 0.f, lse = NEG_INF;
  for (int i = 0; i < LC; ++i) {
    float zp = zpre[idx], dv = dtb[idx];
    float ln_z  = -softplusf(-zp);
    float ln_da = -softplusf(-dv);
    float ln_o  = -softplusf(dv);
    cum += ln_o;
    lse = lae(lse, ln_z + ln_da - cum);
    const float gc = P + cum;                 // global cumsum
    const float gl = lae(R, lse - P);         // global logcumsumexp
    const float hv = __expf(gl + gc) + __expf(gc) * hid;
    outH[idx] = hv;
    h_bf[idx] = f2bf(hv);
    idx += Dd;
  }
}

// ---------------- WMMA bf16 GEMM with fused epilogues ----------------
// C[m,n] = A[m,:] (bf16, row-major [M,K]) . Bw[:,n] (fp32, row-major [K,N])
// Block tile 64M x 256N, 8 waves (2x4), wave tile 32M x 64N, BK=32.
// E=0: outF = acc + bias
// E=1: outF = (acc+bias) * silu(aux1f) + aux2        (y GEMM + residual)
// E=2: outB = bf16( silu(acc+bias) )                 (FFN gate GEMM)
// E=3: outB = bf16( (acc+bias) * bf2f(aux1b) )       (FFN fc GEMM)
// E=4: outF = acc + bias + aux2                      (FFN out GEMM + residual)
template <int E>
__global__ __launch_bounds__(256)
void gemm_bf16(const unsigned short* __restrict__ A, const float* __restrict__ Bw,
               const float* __restrict__ bias,
               float* __restrict__ outF, unsigned short* __restrict__ outB,
               const float* __restrict__ aux1f, const unsigned short* __restrict__ aux1b,
               const float* __restrict__ aux2,
               int N, int K) {
  constexpr int BM = 64, BN = 256, BK = 32, LDP = BK + 4;   // 72B rows: 8B-aligned, 18-bank stride
  __shared__ unsigned short sA[BM][LDP];
  __shared__ unsigned short sB[BN][LDP];                     // transposed [n][k]

  const int m0 = blockIdx.y * BM;
  const int n0 = blockIdx.x * BN;
  const int t = threadIdx.x, lane = t & 31, w = t >> 5;
  const int wm = (w >> 2) * 32, wn = (w & 3) * 64;           // 2x4 wave grid
  const int r = lane & 15, kh = lane >> 4;

  // --- strength-reduced per-thread staging pointers ---
  // A: 64x32 bf16 = 512 uint2; thread t handles rows (t>>3) and (t>>3)+32, cols (t&7)*4
  const int arr = t >> 3, akc = (t & 7) * 4;
  const unsigned short* pA = A + (size_t)(m0 + arr) * K + akc;
  const size_t aRowSkip = (size_t)32 * K;
  // B: 32x256 fp32 = 2048 float4; thread t: rows (t>>6)+4i, cols (t&63)*4
  const int bkk = t >> 6, bnn = (t & 63) * 4;
  const float* pB = Bw + (size_t)bkk * N + (n0 + bnn);
  const size_t bRowSkip = (size_t)4 * N;        // 4 rows per sub-step
  const size_t bStep    = (size_t)BK * N;       // per K-iteration

  union FB { v16bf v; unsigned int u[8]; };
  v8f acc[2][4] = {};

  for (int k0 = 0; k0 < K; k0 += BK) {
    // stage A tile: two b64 loads -> ds_store_b64
#pragma unroll
    for (int i = 0; i < 2; ++i) {
      const uint2 val = *(const uint2*)(pA + i * aRowSkip);
      *(uint2*)&sA[arr + 32 * i][akc] = val;
    }
    // stage B tile: eight b128 loads, convert fp32->bf16, transpose into [n][k]
#pragma unroll
    for (int i = 0; i < 8; ++i) {
      const float4 v4 = *(const float4*)(pB + (size_t)i * bRowSkip);
      const int kk = bkk + 4 * i;
      sB[bnn + 0][kk] = f2bf(v4.x);
      sB[bnn + 1][kk] = f2bf(v4.y);
      sB[bnn + 2][kk] = f2bf(v4.z);
      sB[bnn + 3][kk] = f2bf(v4.w);
    }
    pA += BK;
    if (k0 + BK < K) {   // hint next tiles toward cache (global_prefetch_b8)
      __builtin_prefetch(pA, 0, 3);
      __builtin_prefetch(pB + bStep, 0, 3);
    }
    pB += bStep;
    __syncthreads();

    // assemble fragments per ISA VGPR layouts (wave32)
    FB af[2], bf[4];
#pragma unroll
    for (int s = 0; s < 2; ++s)
#pragma unroll
      for (int p = 0; p < 8; ++p) {
        const int ka = ((p >> 2) * 16) + kh * 8 + ((p & 3) * 2);  // A: interleaved K
        af[s].u[p] = *(const unsigned int*)&sA[wm + s * 16 + r][ka];
      }
#pragma unroll
    for (int s = 0; s < 4; ++s)
#pragma unroll
      for (int p = 0; p < 8; ++p) {
        const int kb = ((p >> 2) * 16) + kh * 8 + ((p & 3) * 2);  // B: interleaved K (matches A)
        bf[s].u[p] = *(const unsigned int*)&sB[wn + s * 16 + r][kb];
      }
#pragma unroll
    for (int i = 0; i < 2; ++i)
#pragma unroll
      for (int j = 0; j < 4; ++j)
        acc[i][j] = __builtin_amdgcn_wmma_f32_16x16x32_bf16(
            false, af[i].v, false, bf[j].v, (short)0, acc[i][j], false, false);
    __syncthreads();
  }

  // epilogue: C/D layout -> m = base + vgpr + 8*(lane>=16), n = base + (lane&15)
  const int ml = kh * 8, nl = r;
#pragma unroll
  for (int i = 0; i < 2; ++i) {
#pragma unroll
    for (int j = 0; j < 4; ++j) {
      const int mb = m0 + wm + i * 16 + ml;
      const int nn = n0 + wn + j * 16 + nl;
      const float bn = bias[nn];
#pragma unroll
      for (int e = 0; e < 8; ++e) {
        const size_t o = (size_t)(mb + e) * N + nn;
        const float v = acc[i][j][e] + bn;
        if constexpr (E == 0) {
          outF[o] = v;
        } else if constexpr (E == 1) {
          outF[o] = v * siluf(aux1f[o]) + aux2[o];
        } else if constexpr (E == 2) {
          outB[o] = f2bf(siluf(v));
        } else if constexpr (E == 3) {
          outB[o] = f2bf(v * bf2f(aux1b[o]));
        } else {
          outF[o] = v + aux2[o];
        }
      }
    }
  }
}

// ---------------- workspace layout (bytes) ----------------
constexpr size_t SZ_BF   = (size_t)Mtot * Dd * 2;        // 32 MB
constexpr size_t SZ_F32  = (size_t)Mtot * Dd * 4;        // 64 MB
constexpr size_t SZ_SC   = (size_t)Bb * Dd * CHUNKS * 4; // 512 KB
constexpr size_t SZ_CH16 = (size_t)CHM * FFf * 2;        // 32 MB
constexpr size_t OFF_XN  = 0;                      // xn_bf -> x2n_bf
constexpr size_t OFF_ZP  = OFF_XN + SZ_BF;         // zpre  -> t2
constexpr size_t OFF_DT  = OFF_ZP + SZ_F32;        // dt    -> x2
constexpr size_t OFF_HB  = OFF_DT + SZ_F32;        // h bf16
constexpr size_t OFF_CS  = OFF_HB + SZ_BF;
constexpr size_t OFF_CL  = OFF_CS + SZ_SC;
constexpr size_t OFF_PE  = OFF_CL + SZ_SC;
constexpr size_t OFF_RE  = OFF_PE + SZ_SC;
constexpr size_t OFF_SI  = OFF_RE + SZ_SC;         // silu buf (chunked FFN)
constexpr size_t OFF_G   = OFF_SI + SZ_CH16;       // gated buf

extern "C" void kernel_launch(void* const* d_in, const int* in_sizes, int n_in,
                              void* d_out, int out_size, void* d_ws, size_t ws_size,
                              hipStream_t stream) {
  (void)in_sizes; (void)n_in; (void)out_size; (void)ws_size;
  const float* x        = (const float*)d_in[0];
  const float* hidden   = (const float*)d_in[1];
  const float* w_ln_z   = (const float*)d_in[2];
  const float* b_ln_z   = (const float*)d_in[3];
  const float* w_dt     = (const float*)d_in[4];
  const float* b_dt     = (const float*)d_in[5];
  const float* w_y      = (const float*)d_in[6];
  const float* b_y      = (const float*)d_in[7];
  const float* w_y_act  = (const float*)d_in[8];
  const float* b_y_act  = (const float*)d_in[9];
  const float* w_fc     = (const float*)d_in[10];
  const float* b_fc     = (const float*)d_in[11];
  const float* w_fc_act = (const float*)d_in[12];
  const float* b_fc_act = (const float*)d_in[13];
  const float* w_out    = (const float*)d_in[14];
  const float* b_out    = (const float*)d_in[15];
  const float* g_sio    = (const float*)d_in[16];
  const float* g_ffn    = (const float*)d_in[17];

  float* outX = (float*)d_out;
  float* outH = outX + (size_t)Mtot * Dd;

  char* ws = (char*)d_ws;
  unsigned short* xn_bf = (unsigned short*)(ws + OFF_XN);
  float*          zpre  = (float*)(ws + OFF_ZP);
  float*          dtb   = (float*)(ws + OFF_DT);
  unsigned short* h_bf  = (unsigned short*)(ws + OFF_HB);
  float*          csum  = (float*)(ws + OFF_CS);
  float*          clse  = (float*)(ws + OFF_CL);
  float*          Pexc  = (float*)(ws + OFF_PE);
  float*          Rexc  = (float*)(ws + OFF_RE);
  unsigned short* silub = (unsigned short*)(ws + OFF_SI);
  unsigned short* gbuf  = (unsigned short*)(ws + OFF_G);

  const dim3 blk(256);
  const dim3 gD(Dd / 256, Mtot / 64);    // D-wide GEMMs over all rows

  // 1) RMSNorm(x) -> bf16
  rmsnorm_bf16_kernel<<<Mtot, blk, 0, stream>>>(x, g_sio, xn_bf);
  // 2) z_pre and dt GEMMs
  gemm_bf16<0><<<gD, blk, 0, stream>>>(xn_bf, w_ln_z, b_ln_z, zpre, nullptr,
                                       nullptr, nullptr, nullptr, Dd, Dd);
  gemm_bf16<0><<<gD, blk, 0, stream>>>(xn_bf, w_dt, b_dt, dtb, nullptr,
                                       nullptr, nullptr, nullptr, Dd, Dd);
  // 3) chunked scan -> h (fp32 to output, bf16 for GEMM)
  scan_p1<<<(Bb * CHUNKS * Dd) / 256, blk, 0, stream>>>(zpre, dtb, csum, clse);
  scan_p2<<<(Bb * Dd) / 256, blk, 0, stream>>>(csum, clse, Pexc, Rexc);
  scan_p3<<<(Bb * CHUNKS * Dd) / 256, blk, 0, stream>>>(zpre, dtb, Pexc, Rexc,
                                                        hidden, outH, h_bf);
  // 4) t2 = xn@w_y_act + b (reuses zpre buffer), then y + residual -> x2 (dt buffer)
  gemm_bf16<0><<<gD, blk, 0, stream>>>(xn_bf, w_y_act, b_y_act, zpre, nullptr,
                                       nullptr, nullptr, nullptr, Dd, Dd);
  gemm_bf16<1><<<gD, blk, 0, stream>>>(h_bf, w_y, b_y, dtb, nullptr,
                                       zpre, nullptr, x, Dd, Dd);
  // 5) RMSNorm(x2) -> bf16 (reuses xn buffer)
  rmsnorm_bf16_kernel<<<Mtot, blk, 0, stream>>>(dtb, g_ffn, xn_bf);
  // 6) SwiGLU FFN in row chunks, residual fused into final GEMM
  for (int mc = 0; mc < Mtot / CHM; ++mc) {
    const unsigned short* a2 = xn_bf + (size_t)mc * CHM * Dd;
    gemm_bf16<2><<<dim3(FFf / 256, CHM / 64), blk, 0, stream>>>(
        a2, w_fc_act, b_fc_act, nullptr, silub, nullptr, nullptr, nullptr, FFf, Dd);
    gemm_bf16<3><<<dim3(FFf / 256, CHM / 64), blk, 0, stream>>>(
        a2, w_fc, b_fc, nullptr, gbuf, nullptr, silub, nullptr, FFf, Dd);
    gemm_bf16<4><<<dim3(Dd / 256, CHM / 64), blk, 0, stream>>>(
        gbuf, w_out, b_out, outX + (size_t)mc * CHM * Dd, nullptr,
        nullptr, nullptr, dtb + (size_t)mc * CHM * Dd, Dd, FFf);
  }
}
